// ResGatedGCNModel_29308856828500
// MI455X (gfx1250) — compile-verified, matrix operand order checked by reading, not run on hardware
//
#include <hip/hip_runtime.h>

#define N_NODES 50000
#define N_EDGES 600000
#define IC      64
#define HDIM    128
#define OC      32
#define BN_EPS  1e-5f

// ---- CDNA5 WMMA types -------------------------------------------------------
typedef __attribute__((ext_vector_type(16))) __bf16       v16bf;
typedef __attribute__((ext_vector_type(8)))  float        v8f;
typedef __attribute__((ext_vector_type(4)))  unsigned int u32x4;
typedef __attribute__((ext_vector_type(2)))  unsigned int u32x2;
typedef __attribute__((ext_vector_type(2)))  __bf16       v2bf;

union Frag { v16bf bf; u32x4 u[2]; };

__device__ __forceinline__ unsigned short f2bf(float f) {
#if __has_builtin(__builtin_amdgcn_cvt_pk_bf16_f32)
  v2bf p = __builtin_amdgcn_cvt_pk_bf16_f32(f, f);
  union { v2bf v; unsigned int u; } cv;
  cv.v = p;
  return (unsigned short)(cv.u & 0xffffu);
#else
  unsigned int u = __float_as_uint(f);
  u += 0x7fffu + ((u >> 16) & 1u);           // round-to-nearest-even
  return (unsigned short)(u >> 16);
#endif
}
__device__ __forceinline__ float bf2f(unsigned short b) {
  return __uint_as_float(((unsigned int)b) << 16);
}

// One wave computes a 16x16 f32 tile of  A[16xK] * Bt[16xK]^T  with bf16 inputs.
// A-fragment layout (CDNA5 ISA, 16-bit A 16x32): lane m=l&15, hi=l>>4,
// VGPR0..3 = K = k0+8*hi .. +7 (consecutive), VGPR4..7 = K = k0+16+8*hi .. +7.
// Bt is stored row-per-output-column ([n][k]) so B loads use the same pattern.
__device__ __forceinline__ v8f wmma_tile(const unsigned short* __restrict__ A, int lda,
                                         const unsigned short* __restrict__ Bt, int ldb,
                                         int K) {
  const int lane = threadIdx.x & 31;
  const int m    = lane & 15;
  const int hi   = lane >> 4;
  v8f c = {0.f, 0.f, 0.f, 0.f, 0.f, 0.f, 0.f, 0.f};
#pragma unroll 2
  for (int k0 = 0; k0 < K; k0 += 32) {
    Frag a, b;
    const unsigned short* ap = A + (size_t)m * lda + k0 + 8 * hi;
    a.u[0] = *(const u32x4*)(ap);
    a.u[1] = *(const u32x4*)(ap + 16);
    const unsigned short* bp = Bt + (size_t)m * ldb + k0 + 8 * hi;
    b.u[0] = *(const u32x4*)(bp);
    b.u[1] = *(const u32x4*)(bp + 16);
    c = __builtin_amdgcn_wmma_f32_16x16x32_bf16(false, a.bf, false, b.bf,
                                                (short)0, c, false, false);
  }
  return c;
}

// ---- weight prep: f32 -> bf16, transposed to [out][in] ----------------------
__global__ void prep_weights_kernel(const float* __restrict__ proj_w,
                                    const float* __restrict__ conv_w,
                                    const float* __restrict__ head_w,
                                    unsigned short* __restrict__ projT,
                                    unsigned short* __restrict__ convT,
                                    unsigned short* __restrict__ headT) {
  int i = blockIdx.x * blockDim.x + threadIdx.x;
  if (i < HDIM * IC) {                                   // projT[n][k], n<128,k<64
    int n = i / IC, k = i % IC;
    projT[i] = f2bf(proj_w[k * HDIM + n]);
  } else if (i < HDIM * IC + 12 * HDIM * HDIM) {         // convT[w][n][k], w = l*4+lin
    int j = i - HDIM * IC;
    int w = j / (HDIM * HDIM);
    int r = j % (HDIM * HDIM);
    int n = r / HDIM, k = r % HDIM;
    convT[j] = f2bf(conv_w[((size_t)(w * HDIM + k)) * HDIM + n]);
  } else if (i < HDIM * IC + 12 * HDIM * HDIM + OC * HDIM) { // headT[n][k]
    int j = i - (HDIM * IC + 12 * HDIM * HDIM);
    int n = j / HDIM, k = j % HDIM;
    headT[j] = f2bf(head_w[k * OC + n]);
  }
}

__global__ void convert_x_kernel(const float* __restrict__ x,
                                 unsigned short* __restrict__ xb) {
  int i = blockIdx.x * blockDim.x + threadIdx.x;
  if (i < N_NODES * IC) xb[i] = f2bf(x[i]);
}

// ---- proj GEMM: h = relu(x @ W + b), K=64, out bf16 -------------------------
// grid = 3125 row tiles, 8 waves -> 8 column tiles covering H=128
__global__ void proj_gemm_kernel(const unsigned short* __restrict__ xb,
                                 const unsigned short* __restrict__ projT,
                                 const float* __restrict__ proj_b,
                                 unsigned short* __restrict__ hbf) {
  const int wave = threadIdx.x >> 5;
  const int lane = threadIdx.x & 31;
  const int rt   = blockIdx.x;
  const unsigned short* A  = xb + (size_t)rt * 16 * IC;
  const unsigned short* Bt = projT + (size_t)wave * 16 * IC;
  v8f c = wmma_tile(A, IC, Bt, IC, IC);
  const int n   = lane & 15;
  const int hi  = lane >> 4;
  const int col = wave * 16 + n;
  const float bias = proj_b[col];
#pragma unroll
  for (int r = 0; r < 8; ++r) {
    int row = rt * 16 + hi * 8 + r;
    float v = fmaxf(c[r] + bias, 0.f);
    hbf[(size_t)row * HDIM + col] = f2bf(v);
  }
}

// ---- fused layer GEMM: k,q,v (bf16) + skip (f32 into accumulator) ----------
// grid = 3125 row tiles, 8 waves loop over 32 column tiles (4 linears x 128).
// The A fragment (node tile) is invariant across column tiles: preload it once
// into registers (8 x u32x4 = 32 VGPRs) and stream only the B fragments.
__global__ void conv_gemm_kernel(const unsigned short* __restrict__ hbf,
                                 const unsigned short* __restrict__ convTl,
                                 const float* __restrict__ conv_b, int layer,
                                 unsigned short* __restrict__ kqv,
                                 float* __restrict__ hnew) {
  const int wave = threadIdx.x >> 5;
  const int lane = threadIdx.x & 31;
  const int rt   = blockIdx.x;
  const int n    = lane & 15;
  const int hi   = lane >> 4;
  const unsigned short* A = hbf + (size_t)rt * 16 * HDIM;
  const size_t sec = (size_t)N_NODES * HDIM;

  Frag a[4];
#pragma unroll
  for (int kk = 0; kk < 4; ++kk) {
    const unsigned short* ap = A + (size_t)n * HDIM + kk * 32 + 8 * hi;
    a[kk].u[0] = *(const u32x4*)(ap);
    a[kk].u[1] = *(const u32x4*)(ap + 16);
  }

  for (int ct = wave; ct < 32; ct += 8) {
    const unsigned short* Bt = convTl + (size_t)ct * 16 * HDIM;
    v8f c = {0.f, 0.f, 0.f, 0.f, 0.f, 0.f, 0.f, 0.f};
#pragma unroll
    for (int kk = 0; kk < 4; ++kk) {
      Frag b;
      const unsigned short* bp = Bt + (size_t)n * HDIM + kk * 32 + 8 * hi;
      b.u[0] = *(const u32x4*)(bp);
      b.u[1] = *(const u32x4*)(bp + 16);
      c = __builtin_amdgcn_wmma_f32_16x16x32_bf16(false, a[kk].bf, false, b.bf,
                                                  (short)0, c, false, false);
    }
    const int j  = ct >> 3;                   // 0=key 1=query 2=value 3=skip
    const int cl = (ct & 7) * 16 + n;         // column within this linear
    const float bias = conv_b[(size_t)(layer * 4 + j) * HDIM + cl];
    if (j < 3) {
      unsigned short* dst = kqv + (size_t)j * sec;
#pragma unroll
      for (int r = 0; r < 8; ++r) {
        int row = rt * 16 + hi * 8 + r;
        dst[(size_t)row * HDIM + cl] = f2bf(c[r] + bias);
      }
    } else {
#pragma unroll
      for (int r = 0; r < 8; ++r) {
        int row = rt * 16 + hi * 8 + r;
        hnew[(size_t)row * HDIM + cl] = c[r] + bias;   // skip term seeds agg
      }
    }
  }
}

// ---- edge kernel: hnew[dst] += sigmoid(k[dst]+q[src]) * v[src] --------------
// one wave per edge; lane covers 4 channels (8-byte bf16 gathers, f32 atomics)
__global__ void edge_kernel(const int* __restrict__ ei,
                            const unsigned short* __restrict__ kqv,
                            float* __restrict__ hnew) {
  const int lane = threadIdx.x & 31;
  const int e    = blockIdx.x * (blockDim.x >> 5) + (threadIdx.x >> 5);
  if (e >= N_EDGES) return;
  const int s = ei[e];
  const int d = ei[N_EDGES + e];
  const size_t sec = (size_t)N_NODES * HDIM;
  const int c0 = lane * 4;
  u32x2 ku = *(const u32x2*)(kqv + (size_t)d * HDIM + c0);
  u32x2 qu = *(const u32x2*)(kqv + sec + (size_t)s * HDIM + c0);
  u32x2 vu = *(const u32x2*)(kqv + 2 * sec + (size_t)s * HDIM + c0);
  float* out = hnew + (size_t)d * HDIM + c0;
#pragma unroll
  for (int i = 0; i < 4; ++i) {
    unsigned int kw = (i < 2) ? ku.x : ku.y;
    unsigned int qw = (i < 2) ? qu.x : qu.y;
    unsigned int vw = (i < 2) ? vu.x : vu.y;
    int sh = (i & 1) ? 16 : 0;
    float kf = bf2f((unsigned short)(kw >> sh));
    float qf = bf2f((unsigned short)(qw >> sh));
    float vf = bf2f((unsigned short)(vw >> sh));
    float g  = 1.f / (1.f + __expf(-(kf + qf)));
    atomicAdd(&out[i], g * vf);
  }
}

// ---- batchnorm stats: per-channel sum / sumsq -------------------------------
__global__ void zero_stats_kernel(float* __restrict__ stats) {
  stats[threadIdx.x] = 0.f;      // 256 entries: [0..127]=sum [128..255]=sumsq
}

__global__ void bn_stats_kernel(const float* __restrict__ h,
                                float* __restrict__ stats) {
  __shared__ float ssum[256];
  __shared__ float ssq[256];
  const int tid = threadIdx.x;
  const size_t total  = (size_t)N_NODES * HDIM;
  const size_t stride = (size_t)gridDim.x * blockDim.x;   // multiple of 128
  float s = 0.f, s2 = 0.f;
  for (size_t i = (size_t)blockIdx.x * blockDim.x + tid; i < total; i += stride) {
    float v = h[i];
    s += v; s2 += v * v;
  }
  ssum[tid] = s; ssq[tid] = s2;
  __syncthreads();
  if (tid < 128) {
    int ch = tid;                       // (blockIdx*256+tid)%128 == tid%128
    atomicAdd(&stats[ch],        ssum[tid] + ssum[tid + 128]);
    atomicAdd(&stats[128 + ch],  ssq[tid]  + ssq[tid + 128]);
  }
}

// ---- batchnorm apply + relu -> bf16 h for next layer ------------------------
__global__ void bn_apply_kernel(const float* __restrict__ hnew,
                                const float* __restrict__ stats,
                                const float* __restrict__ gamma,
                                const float* __restrict__ beta,
                                unsigned short* __restrict__ hbf) {
  const size_t idx = (size_t)blockIdx.x * blockDim.x + threadIdx.x;
  if (idx >= (size_t)N_NODES * HDIM) return;
  const int ch = (int)(idx & (HDIM - 1));
  const float inv_n = 1.f / (float)N_NODES;
  float mean = stats[ch] * inv_n;
  float var  = stats[128 + ch] * inv_n - mean * mean;
  float y = gamma[ch] * (hnew[idx] - mean) * rsqrtf(var + BN_EPS) + beta[ch];
  hbf[idx] = f2bf(fmaxf(y, 0.f));
}

// ---- head GEMM: out = h @ Wh + bh, K=128, 32 cols, f32 out ------------------
__global__ void head_gemm_kernel(const unsigned short* __restrict__ hbf,
                                 const unsigned short* __restrict__ headT,
                                 const float* __restrict__ head_b,
                                 float* __restrict__ out) {
  const int wave = threadIdx.x >> 5;
  const int lane = threadIdx.x & 31;
  const int g    = blockIdx.x * 8 + wave;      // wave-tile id
  if (g >= 3125 * 2) return;                   // whole wave exits together
  const int rt = g >> 1;
  const int ct = g & 1;
  const unsigned short* A  = hbf + (size_t)rt * 16 * HDIM;
  const unsigned short* Bt = headT + (size_t)ct * 16 * HDIM;
  v8f c = wmma_tile(A, HDIM, Bt, HDIM, HDIM);
  const int n   = lane & 15;
  const int hi  = lane >> 4;
  const int col = ct * 16 + n;
  const float bias = head_b[col];
#pragma unroll
  for (int r = 0; r < 8; ++r) {
    int row = rt * 16 + hi * 8 + r;
    out[(size_t)row * OC + col] = c[r] + bias;
  }
}

// ---- launcher ---------------------------------------------------------------
extern "C" void kernel_launch(void* const* d_in, const int* in_sizes, int n_in,
                              void* d_out, int out_size, void* d_ws, size_t ws_size,
                              hipStream_t stream) {
  const float* x        = (const float*)d_in[0];
  const int*   ei       = (const int*)d_in[1];
  const float* proj_w   = (const float*)d_in[2];
  const float* proj_b   = (const float*)d_in[3];
  const float* conv_w   = (const float*)d_in[4];
  const float* conv_b   = (const float*)d_in[5];
  const float* bn_gamma = (const float*)d_in[6];
  const float* bn_beta  = (const float*)d_in[7];
  const float* head_w   = (const float*)d_in[8];
  const float* head_b   = (const float*)d_in[9];
  float* out = (float*)d_out;

  // workspace carve-out (~84 MB), 256B aligned
  char* p = (char*)d_ws;
  auto carve = [&p](size_t bytes) {
    char* r = p;
    p += (bytes + 255) & ~(size_t)255;
    return r;
  };
  unsigned short* xb    = (unsigned short*)carve((size_t)N_NODES * IC * 2);
  unsigned short* hbf   = (unsigned short*)carve((size_t)N_NODES * HDIM * 2);
  float*          hnew  = (float*)carve((size_t)N_NODES * HDIM * 4);
  unsigned short* kqv   = (unsigned short*)carve((size_t)3 * N_NODES * HDIM * 2);
  unsigned short* projT = (unsigned short*)carve((size_t)HDIM * IC * 2);
  unsigned short* convT = (unsigned short*)carve((size_t)12 * HDIM * HDIM * 2);
  unsigned short* headT = (unsigned short*)carve((size_t)OC * HDIM * 2);
  float*          stats = (float*)carve(256 * 4);

  const int BLK = 256;
  // weights + input to bf16 (deterministic each call)
  {
    int total = HDIM * IC + 12 * HDIM * HDIM + OC * HDIM;  // 208896
    prep_weights_kernel<<<(total + BLK - 1) / BLK, BLK, 0, stream>>>(
        proj_w, conv_w, head_w, projT, convT, headT);
    convert_x_kernel<<<(N_NODES * IC + BLK - 1) / BLK, BLK, 0, stream>>>(x, xb);
  }

  // h = relu(x @ proj_w + proj_b)
  proj_gemm_kernel<<<N_NODES / 16, BLK, 0, stream>>>(xb, projT, proj_b, hbf);

  for (int l = 0; l < 3; ++l) {
    conv_gemm_kernel<<<N_NODES / 16, BLK, 0, stream>>>(
        hbf, convT + (size_t)l * 4 * HDIM * HDIM, conv_b, l, kqv, hnew);
    edge_kernel<<<N_EDGES / 8, BLK, 0, stream>>>(ei, kqv, hnew);
    zero_stats_kernel<<<1, 256, 0, stream>>>(stats);
    bn_stats_kernel<<<512, BLK, 0, stream>>>(hnew, stats);
    bn_apply_kernel<<<(N_NODES * HDIM) / BLK, BLK, 0, stream>>>(
        hnew, stats, bn_gamma + (size_t)l * HDIM, bn_beta + (size_t)l * HDIM, hbf);
  }

  head_gemm_kernel<<<(3125 * 2 + 7) / 8, BLK, 0, stream>>>(hbf, headT, head_b, out);
}
